// Evoformer_66984309948702
// MI455X (gfx1250) — matrix operand order repaired and checked by reading, not compiled, not run on hardware
//
#include <hip/hip_runtime.h>

typedef unsigned short u16;
typedef unsigned int   u32;
typedef __attribute__((ext_vector_type(16))) __bf16 v16bf;
typedef __attribute__((ext_vector_type(8)))  float  v8f;
typedef __attribute__((ext_vector_type(4)))  u32    v4u;

union BF16x16 { v16bf v; u16 s[16]; u32 u[8]; v4u q[2]; };

__device__ __forceinline__ u16 f2bf(float f) {
    u32 u = __builtin_bit_cast(u32, f);
    u32 r = u + 0x7FFFu + ((u >> 16) & 1u);
    return (u16)(r >> 16);
}

// ---------------------------------------------------------------------------
// Weight prep: f32 [K,N] -> bf16 transposed+padded [Npad, Kpad]
// ---------------------------------------------------------------------------
__global__ void prep_weight_kernel(const float* __restrict__ W, u16* __restrict__ Wt,
                                   int K, int N, int Kpad, int Npad) {
    long idx = (long)blockIdx.x * 256 + threadIdx.x;
    long tot = (long)Kpad * Npad;
    if (idx >= tot) return;
    int k = (int)(idx % Kpad);
    int n = (int)(idx / Kpad);
    u16 v = 0;
    if (n < N && k < K) v = f2bf(W[(long)k * N + n]);
    Wt[idx] = v;
}

// ---------------------------------------------------------------------------
// LayerNorm f32 [R,D] -> bf16 [R,D].  blockDim = D (128 or 256), grid = R.
// ---------------------------------------------------------------------------
__global__ void ln_bf16_kernel(const float* __restrict__ X, const float* __restrict__ G,
                               const float* __restrict__ B, u16* __restrict__ Y, int D) {
    __shared__ float sred[8];
    __shared__ float sbc[2];
    const long r = blockIdx.x;
    const int t = threadIdx.x;
    const int lane = t & 31, w = t >> 5, nw = blockDim.x >> 5;
    float x = X[r * D + t];
    float s = x;
    for (int o = 16; o; o >>= 1) s += __shfl_xor(s, o, 32);
    if (lane == 0) sred[w] = s;
    __syncthreads();
    if (t == 0) { float tot = 0.f; for (int i = 0; i < nw; i++) tot += sred[i]; sbc[0] = tot / D; }
    __syncthreads();
    float mean = sbc[0];
    float d = x - mean;
    float v2 = d * d;
    for (int o = 16; o; o >>= 1) v2 += __shfl_xor(v2, o, 32);
    __syncthreads();
    if (lane == 0) sred[w] = v2;
    __syncthreads();
    if (t == 0) { float tot = 0.f; for (int i = 0; i < nw; i++) tot += sred[i]; sbc[1] = tot / D; }
    __syncthreads();
    float var = sbc[1];
    float y = d * rsqrtf(var + 1e-5f) * G[t] + B[t];
    Y[r * D + t] = f2bf(y);
}

// ---------------------------------------------------------------------------
// bf16 WMMA GEMM: C[M,N] = act(A[M,K] @ Wt[N,K]^T + bias) (+residual add)
// A bf16 row-major (lda), Wt bf16 row-major [Npad, ldw=Kpad].
// Block = 256 threads = 8 waves; each wave computes NT 16x16 tiles sharing A.
// OUTMODE: 0 = f32 row-major (opt. +=), 1 = bf16 row-major, 2 = bf16 col-major
// ---------------------------------------------------------------------------
template <int NT, int ACT, int OUTMODE, int ADD>
__global__ void gemm_kernel(const u16* __restrict__ A, int lda,
                            const u16* __restrict__ Wt, int ldw,
                            const float* __restrict__ bias,
                            float* __restrict__ Cf, u16* __restrict__ Cb, int ldc,
                            int M, int Ntrue, int K) {
    const int lane = threadIdx.x & 31;
    const int wave = threadIdx.x >> 5;
    const int mt = blockIdx.x * 8 + wave;
    if (mt * 16 >= M) return;
    const int n0 = blockIdx.y * (16 * NT);
    const int row = lane & 15;
    const bool hi = lane >= 16;

    const u16* ap = A + (long)(mt * 16 + row) * lda + (hi ? 8 : 0);
    const u16* bp0 = Wt + (long)(n0 + row) * ldw + (hi ? 8 : 0);

    v8f acc[NT];
    for (int t = 0; t < NT; t++)
        for (int i = 0; i < 8; i++) acc[t][i] = 0.f;

    for (int k0 = 0; k0 < K; k0 += 32) {
        BF16x16 a;
        a.q[0] = *(const v4u*)(ap + k0);
        a.q[1] = *(const v4u*)(ap + k0 + 16);
        __builtin_prefetch(ap + k0 + 64, 0, 0);
#pragma unroll
        for (int t = 0; t < NT; t++) {
            BF16x16 b;
            const u16* bp = bp0 + (long)t * 16 * ldw + k0;
            b.q[0] = *(const v4u*)(bp);
            b.q[1] = *(const v4u*)(bp + 16);
            acc[t] = __builtin_amdgcn_wmma_f32_16x16x32_bf16(
                false, a.v, false, b.v, (short)0, acc[t], false, false);
        }
    }

#pragma unroll
    for (int t = 0; t < NT; t++) {
        int col = n0 + t * 16 + row;
        if (col >= Ntrue) continue;
        float bv = bias ? bias[col] : 0.f;
#pragma unroll
        for (int vi = 0; vi < 8; vi++) {
            int r = mt * 16 + vi + (hi ? 8 : 0);
            float val = acc[t][vi] + bv;
            if (ACT == 1) val = fmaxf(val, 0.f);
            else if (ACT == 2) val = 1.f / (1.f + __expf(-val));
            if (OUTMODE == 1) Cb[(long)r * ldc + col] = f2bf(val);
            else if (OUTMODE == 2) Cb[(long)col * M + r] = f2bf(val);
            else if (ADD) Cf[(long)r * ldc + col] += val;
            else Cf[(long)r * ldc + col] = val;
        }
    }
}

// ---------------------------------------------------------------------------
// Stride-generic attention. One wave per (qtile, head, batch).
// scores[16,S] via WMMA (head dim C=8 zero-padded into K=32), f32 softmax in
// LDS, probs->bf16 in LDS, then either apply (probs@V, gate, bf16 out) or dump
// probs to global (for the crossed terminal attention).
// V is COLUMN-MAJOR: Vt[(h*C+c)*vM + b*S + k]  -> contiguous b128 B-fragments.
// Apply-phase B columns c>=C carry garbage (valid clamped loads); harmless
// since those D columns are never stored -> no EXEC branching in inner loops.
// ---------------------------------------------------------------------------
template <int WRITE_PROBS>
__global__ void attn_kernel(const u16* __restrict__ Q, const u16* __restrict__ K,
                            const u16* __restrict__ Vt, const float* __restrict__ Bias,
                            const float* __restrict__ Gate, u16* __restrict__ Out,
                            u16* __restrict__ Probs, int S, int C, int vM,
                            long q_bs, long q_ss, long q_hs,
                            long k_bs, long k_ss, long k_hs,
                            long bias_qs, long bias_ks, long bias_hs,
                            long g_bs, long g_ss, long g_hs,
                            long o_bs, long o_ss, long o_hs,
                            float scale, int head_ofs) {
    __shared__ float sc[16][256];
    __shared__ u16 pr[16][256];
    const int lane = threadIdx.x & 31;
    const int row = lane & 15;
    const bool hi = lane >= 16;
    const int m0 = blockIdx.x * 16;
    const int h = blockIdx.y + head_ofs;
    const int b = blockIdx.z;
    const v4u zero4 = {0u, 0u, 0u, 0u};

    // Q fragment (A): lanes 0-15 hold c=0..7 in halves 0..7; K-padding -> 0.
    // Unconditional load + vector select (v_cndmask) instead of exec branch.
    BF16x16 aq;
    {
        const u16* qp = Q + (long)b * q_bs + (long)(m0 + row) * q_ss + (long)h * q_hs;
        v4u qv = *(const v4u*)qp;
        aq.q[0] = hi ? zero4 : qv;
        aq.q[1] = zero4;
    }

    const int ntile = S >> 4;
    for (int kt = 0; kt < ntile; ++kt) {
        BF16x16 bk;
        {
            const u16* kp = K + (long)b * k_bs + (long)(kt * 16 + row) * k_ss + (long)h * k_hs;
            v4u kv = *(const v4u*)kp;
            bk.q[0] = hi ? zero4 : kv;
            bk.q[1] = zero4;
        }
        v8f c;
        for (int i = 0; i < 8; i++) c[i] = 0.f;
        c = __builtin_amdgcn_wmma_f32_16x16x32_bf16(false, aq.v, false, bk.v, (short)0, c,
                                                    false, false);
        const int kcol = kt * 16 + row;
#pragma unroll
        for (int vi = 0; vi < 8; vi++) {
            int m = vi + (hi ? 8 : 0);
            int qg = m0 + m;
            sc[m][kcol] = c[vi] * scale +
                          Bias[(long)qg * bias_qs + (long)kcol * bias_ks + (long)h * bias_hs];
        }
    }
    __syncthreads();

    for (int r = 0; r < 16; ++r) {
        float mx = -3.0e38f;
        for (int cix = lane; cix < S; cix += 32) mx = fmaxf(mx, sc[r][cix]);
        for (int o = 16; o; o >>= 1) mx = fmaxf(mx, __shfl_xor(mx, o, 32));
        float sum = 0.f;
        for (int cix = lane; cix < S; cix += 32) {
            float e = __expf(sc[r][cix] - mx);
            sc[r][cix] = e;
            sum += e;
        }
        for (int o = 16; o; o >>= 1) sum += __shfl_xor(sum, o, 32);
        float inv = 1.f / sum;
        for (int cix = lane; cix < S; cix += 32) pr[r][cix] = f2bf(sc[r][cix] * inv);
    }
    __syncthreads();

    if (WRITE_PROBS) {
        for (int r = 0; r < 16; ++r)
            for (int cix = lane; cix < S; cix += 32)
                Probs[((long)b * S + (m0 + r)) * S + cix] = pr[r][cix];
        return;
    }

    // out[16, C] = probs[16,S] @ V[S,C]; V column index clamped (C power of 2)
    v8f co;
    for (int i = 0; i < 8; i++) co[i] = 0.f;
    const u16* vbase = Vt + ((long)h * C + (row & (C - 1))) * vM + (long)b * S;
    for (int k0 = 0; k0 < S; k0 += 32) {
        BF16x16 ap;
        const int kA = k0 + (hi ? 8 : 0);
        ap.q[0] = *(const v4u*)&pr[row][kA];
        ap.q[1] = *(const v4u*)&pr[row][kA + 16];
        BF16x16 bv;
        const int kB = k0 + (hi ? 8 : 0);
        bv.q[0] = *(const v4u*)(vbase + kB);
        bv.q[1] = *(const v4u*)(vbase + kB + 16);
        co = __builtin_amdgcn_wmma_f32_16x16x32_bf16(false, ap.v, false, bv.v, (short)0, co,
                                                     false, false);
    }
    if (row < C) {
#pragma unroll
        for (int vi = 0; vi < 8; vi++) {
            int m = m0 + vi + (hi ? 8 : 0);
            float val = co[vi] * Gate[(long)b * g_bs + (long)m * g_ss + (long)h * g_hs + row];
            Out[(long)b * o_bs + (long)m * o_ss + (long)h * o_hs + row] = f2bf(val);
        }
    }
}

// ---------------------------------------------------------------------------
// Terminal-attention apply:  out[b,l,h,c] = sum_k probs[b,l,k] * V[k,l,h,c]
// One 256-thread block per column l: stage V[:,l,h,:] (S x C bf16) into LDS
// once, then 8 waves each handle 2 b-tiles via WMMA reading b128 from LDS.
// Vt is column-major: Vt[(h*C+c)*vM + k*S + l].   (S == 256 assumed)
// ---------------------------------------------------------------------------
__global__ void attn_apply_t_kernel(const u16* __restrict__ Probs, const u16* __restrict__ Vt,
                                    const float* __restrict__ Gate, u16* __restrict__ Out,
                                    int S, int HC, int C, int h, int vM) {
    __shared__ u16 vds[8][256];
    const int tid = threadIdx.x;
    const int lane = tid & 31;
    const int wave = tid >> 5;
    const int row = lane & 15;
    const bool hi = lane >= 16;
    const int l = blockIdx.x;

    for (int i = tid; i < C * 256; i += blockDim.x) {
        int c = i >> 8;
        int k = i & 255;
        vds[c][k] = Vt[((long)h * C + c) * vM + (long)k * S + l];
    }
    __syncthreads();

    const u16* vrow = &vds[row & (C - 1)][0]; // clamped: c>=C columns never stored
    for (int bt = wave; bt < 16; bt += 8) {
        const int b0 = bt * 16;
        v8f co;
        for (int i = 0; i < 8; i++) co[i] = 0.f;
        const u16* arow = Probs + ((long)(b0 + row) * S + l) * S + (hi ? 8 : 0);
        for (int k0 = 0; k0 < S; k0 += 32) {
            BF16x16 ap;
            ap.q[0] = *(const v4u*)(arow + k0);
            ap.q[1] = *(const v4u*)(arow + k0 + 16);
            BF16x16 bv;
            const int kB = k0 + (hi ? 8 : 0);
            bv.q[0] = *(const v4u*)(vrow + kB);
            bv.q[1] = *(const v4u*)(vrow + kB + 16);
            co = __builtin_amdgcn_wmma_f32_16x16x32_bf16(false, ap.v, false, bv.v, (short)0,
                                                         co, false, false);
        }
        if (row < C) {
#pragma unroll
            for (int vi = 0; vi < 8; vi++) {
                int b = b0 + vi + (hi ? 8 : 0);
                long o = ((long)b * S + l) * HC + h * C + row;
                Out[o] = f2bf(co[vi] * Gate[o]);
            }
        }
    }
}

// ---------------------------------------------------------------------------
// OPM helpers
// ---------------------------------------------------------------------------
// At[(i*Cc+a), n] = Af[(n*S+i)*ldin + a]  (bf16)
__global__ void opm_transpose_kernel(const float* __restrict__ Af, u16* __restrict__ At,
                                     int Nseq, int S, int Cc, int ldin) {
    long idx = (long)blockIdx.x * 256 + threadIdx.x;
    long tot = (long)S * Cc * Nseq;
    if (idx >= tot) return;
    int n = (int)(idx % Nseq);
    long r = idx / Nseq;
    int a = (int)(r % Cc);
    int i = (int)(r / Cc);
    At[idx] = f2bf(Af[((long)n * S + i) * ldin + a]);
}

// X[(i*S+j)*Kpad + a*Cc+b] = O[(i*Cc+a)*(S*Cc) + j*Cc+b] * invN, pad -> 0
__global__ void opm_reorder_kernel(const float* __restrict__ O, u16* __restrict__ X,
                                   int S, int Cc, int Kpad, float invN) {
    long idx = (long)blockIdx.x * 256 + threadIdx.x;
    long tot = (long)S * S * Kpad;
    if (idx >= tot) return;
    int c = (int)(idx % Kpad);
    long rij = idx / Kpad;
    int j = (int)(rij % S);
    int i = (int)(rij / S);
    u16 v = 0;
    if (c < Cc * Cc) {
        int a = c / Cc, bb = c % Cc;
        v = f2bf(O[((long)i * Cc + a) * ((long)S * Cc) + (long)j * Cc + bb] * invN);
    }
    X[idx] = v;
}

// ---------------------------------------------------------------------------
// Host side
// ---------------------------------------------------------------------------
static void launch_gemm(hipStream_t s, int NT, int ACT, int OUTMODE, int ADD,
                        const u16* A, int lda, const u16* Wt, int ldw, const float* bias,
                        float* Cf, u16* Cb, int ldc, int M, int Ntrue, int Npad, int K) {
    dim3 g((M / 16 + 7) / 8, Npad / (16 * NT));
    dim3 b(256);
#define GEMM_CASE(nt, a, om, ad)                                                              \
    if (NT == nt && ACT == a && OUTMODE == om && ADD == ad) {                                 \
        gemm_kernel<nt, a, om, ad><<<g, b, 0, s>>>(A, lda, Wt, ldw, bias, Cf, Cb, ldc, M,     \
                                                   Ntrue, K);                                 \
        return;                                                                               \
    }
    GEMM_CASE(1, 0, 0, 0)
    GEMM_CASE(2, 0, 1, 0)
    GEMM_CASE(2, 0, 2, 0)
    GEMM_CASE(2, 2, 0, 0)
    GEMM_CASE(4, 0, 1, 0)
    GEMM_CASE(4, 0, 2, 0)
    GEMM_CASE(4, 2, 0, 0)
    GEMM_CASE(4, 0, 0, 0)
    GEMM_CASE(4, 0, 0, 1)
    GEMM_CASE(4, 1, 1, 0)
#undef GEMM_CASE
}

extern "C" void kernel_launch(void* const* d_in, const int* in_sizes, int n_in,
                              void* d_out, int out_size, void* d_ws, size_t ws_size,
                              hipStream_t stream) {
    (void)in_sizes; (void)n_in; (void)out_size; (void)ws_size;

    const int NSEQ = 128, L = 256, MD = 256, PD = 128;
    const int MH = 8, MC = 8, PH = 4, PC = 8, OC = 12, EXP = 2;
    const long MR = (long)NSEQ * L;      // 32768 msa rows
    const long PR = (long)L * L;         // 65536 pair rows
    const float iscale = 0.35355339059f; // 1/sqrt(8)

    const float* in_msa = (const float*)d_in[0];
    const float* in_pair = (const float*)d_in[1];
    float* outMsa = (float*)d_out;
    float* outPair = outMsa + MR * MD;

    // ---- workspace bump allocator ----
    char* wsb = (char*)d_ws;
    size_t off = 0;
    auto alloc = [&](size_t bytes) -> void* {
        void* p = wsb + off;
        off = (off + bytes + 255) & ~(size_t)255;
        return p;
    };
    auto au16 = [&](long elems) { return (u16*)alloc((size_t)elems * 2); };
    auto af32 = [&](long elems) { return (float*)alloc((size_t)elems * 4); };

    // weights (bf16, transposed [Npad,Kpad])
    u16* wtq_m = au16(64 * 256); u16* wtk_m = au16(64 * 256);
    u16* wtv_m = au16(64 * 256); u16* wtg_m = au16(64 * 256);
    u16* wtz = au16(16 * 128);   u16* wto_m = au16(256 * 64);
    u16* wt1_m = au16(512 * 256); u16* wt2_m = au16(256 * 512);
    u16* wta = au16(16 * 256);   u16* wtb_o = au16(16 * 256);
    u16* wto_opm = au16(128 * 160);
    u16* wtq_i = au16(32 * 128); u16* wtk_i = au16(32 * 128);
    u16* wtv_i = au16(32 * 128); u16* wtg_i = au16(32 * 128);
    u16* wtb_i = au16(16 * 128); u16* wto_i = au16(128 * 32);
    u16* wtq_t = au16(32 * 128); u16* wtk_t = au16(32 * 128);
    u16* wtv_t = au16(32 * 128); u16* wtg_t = au16(32 * 128);
    u16* wtb_t = au16(16 * 128); u16* wto_t = au16(128 * 32);
    u16* wt1_p = au16(256 * 128); u16* wt2_p = au16(128 * 256);

    // activations (regions reused across stream-ordered stages)
    u16* msa_ln = au16(MR * MD);          // also trans/opm LN
    u16* pair_ln = au16(PR * PD);         // wz LN + later pair LNs
    u16* qb = au16(MR * 64);              // == PR*32
    u16* kb = au16(MR * 64);
    u16* vb = au16(MR * 64);              // column-major [HC][rows]
    float* gate_f = af32(MR * 64);
    float* bias_f = af32(PR * 8);
    u16* attn_o = au16(MR * 64);
    u16* h1 = au16(MR * 512);             // also opm X [PR,160]
    float* a_f = af32(MR * OC);
    float* b_f = af32(MR * OC);
    u16* aT = au16((long)L * OC * NSEQ);
    u16* bT = au16((long)L * OC * NSEQ);
    float* Obuf = af32((long)L * OC * L * OC); // 3072x3072; reused as probs (bf16)
    u16* probs = (u16*)Obuf;

    auto prep = [&](int idx, u16* Wt, int K, int N, int Kpad, int Npad) {
        long tot = (long)Kpad * Npad;
        prep_weight_kernel<<<(tot + 255) / 256, 256, 0, stream>>>(
            (const float*)d_in[idx], Wt, K, N, Kpad, Npad);
    };
    // ---- weight prep ----
    prep(4, wtq_m, 256, 64, 256, 64);  prep(5, wtk_m, 256, 64, 256, 64);
    prep(6, wtv_m, 256, 64, 256, 64);  prep(10, wtg_m, 256, 64, 256, 64);
    prep(9, wtz, 128, 8, 128, 16);     prep(12, wto_m, 64, 256, 64, 256);
    prep(16, wt1_m, 256, 512, 256, 512); prep(18, wt2_m, 512, 256, 512, 256);
    prep(22, wta, 256, 12, 256, 16);   prep(24, wtb_o, 256, 12, 256, 16);
    prep(26, wto_opm, 144, 128, 160, 128);
    prep(30, wtq_i, 128, 32, 128, 32); prep(32, wtk_i, 128, 32, 128, 32);
    prep(34, wtv_i, 128, 32, 128, 32); prep(38, wtg_i, 128, 32, 128, 32);
    prep(36, wtb_i, 128, 4, 128, 16);  prep(40, wto_i, 32, 128, 32, 128);
    prep(44, wtq_t, 128, 32, 128, 32); prep(46, wtk_t, 128, 32, 128, 32);
    prep(48, wtv_t, 128, 32, 128, 32); prep(52, wtg_t, 128, 32, 128, 32);
    prep(50, wtb_t, 128, 4, 128, 16);  prep(54, wto_t, 32, 128, 32, 128);
    prep(58, wt1_p, 128, 256, 128, 256); prep(60, wt2_p, 256, 128, 256, 128);

    // residual init
    hipMemcpyAsync(outMsa, in_msa, (size_t)MR * MD * 4, hipMemcpyDeviceToDevice, stream);
    hipMemcpyAsync(outPair, in_pair, (size_t)PR * PD * 4, hipMemcpyDeviceToDevice, stream);

    // ================= Stage 1: MSA row attention =================
    ln_bf16_kernel<<<(int)MR, MD, 0, stream>>>(in_msa, (const float*)d_in[2],
                                               (const float*)d_in[3], msa_ln, MD);
    ln_bf16_kernel<<<(int)PR, PD, 0, stream>>>(in_pair, (const float*)d_in[7],
                                               (const float*)d_in[8], pair_ln, PD);
    launch_gemm(stream, 4, 0, 1, 0, msa_ln, MD, wtq_m, MD, nullptr, nullptr, qb, 64, MR, 64, 64, MD);
    launch_gemm(stream, 4, 0, 1, 0, msa_ln, MD, wtk_m, MD, nullptr, nullptr, kb, 64, MR, 64, 64, MD);
    launch_gemm(stream, 4, 0, 2, 0, msa_ln, MD, wtv_m, MD, nullptr, nullptr, vb, 64, MR, 64, 64, MD);
    launch_gemm(stream, 4, 2, 0, 0, msa_ln, MD, wtg_m, MD, (const float*)d_in[11], gate_f,
                nullptr, 64, MR, 64, 64, MD);
    launch_gemm(stream, 1, 0, 0, 0, pair_ln, PD, wtz, PD, nullptr, bias_f, nullptr, 8, PR, 8, 16, PD);
    {
        dim3 g(L / 16, MH, NSEQ);
        attn_kernel<0><<<g, 32, 0, stream>>>(
            qb, kb, vb, bias_f, gate_f, attn_o, nullptr, L, MC, (int)MR,
            (long)L * 64, 64, 8,       // q
            (long)L * 64, 64, 8,       // k
            (long)L * MH, MH, 1,       // bias [Lq,Lk,H]
            (long)L * 64, 64, 8,       // gate
            (long)L * 64, 64, 8,       // out
            iscale, 0);
    }
    launch_gemm(stream, 4, 0, 0, 1, attn_o, 64, wto_m, 64, (const float*)d_in[13], outMsa,
                nullptr, MD, MR, MD, MD, 64);

    // ================= Stage 2: MSA transition =================
    ln_bf16_kernel<<<(int)MR, MD, 0, stream>>>(outMsa, (const float*)d_in[14],
                                               (const float*)d_in[15], msa_ln, MD);
    launch_gemm(stream, 4, 1, 1, 0, msa_ln, MD, wt1_m, MD, (const float*)d_in[17], nullptr, h1,
                EXP * MD, MR, EXP * MD, EXP * MD, MD);
    launch_gemm(stream, 4, 0, 0, 1, h1, EXP * MD, wt2_m, EXP * MD, (const float*)d_in[19],
                outMsa, nullptr, MD, MR, MD, MD, EXP * MD);

    // ================= Stage 3: outer product mean =================
    ln_bf16_kernel<<<(int)MR, MD, 0, stream>>>(outMsa, (const float*)d_in[20],
                                               (const float*)d_in[21], msa_ln, MD);
    launch_gemm(stream, 1, 0, 0, 0, msa_ln, MD, wta, MD, (const float*)d_in[23], a_f, nullptr,
                OC, MR, OC, 16, MD);
    launch_gemm(stream, 1, 0, 0, 0, msa_ln, MD, wtb_o, MD, (const float*)d_in[25], b_f, nullptr,
                OC, MR, OC, 16, MD);
    {
        long tot = (long)L * OC * NSEQ;
        opm_transpose_kernel<<<(tot + 255) / 256, 256, 0, stream>>>(a_f, aT, NSEQ, L, OC, OC);
        opm_transpose_kernel<<<(tot + 255) / 256, 256, 0, stream>>>(b_f, bT, NSEQ, L, OC, OC);
    }
    launch_gemm(stream, 4, 0, 0, 0, aT, NSEQ, bT, NSEQ, nullptr, Obuf, nullptr, L * OC,
                L * OC, L * OC, L * OC, NSEQ);
    {
        u16* X = h1; // reuse
        long tot = (long)PR * 160;
        opm_reorder_kernel<<<(tot + 255) / 256, 256, 0, stream>>>(Obuf, X, L, OC, 160,
                                                                  1.0f / NSEQ);
        launch_gemm(stream, 4, 0, 0, 1, X, 160, wto_opm, 160, (const float*)d_in[27], outPair,
                    nullptr, PD, PR, PD, PD, 160);
    }

    // ================= Stage 4: pair attention (initial / row-wise) =========
    ln_bf16_kernel<<<(int)PR, PD, 0, stream>>>(outPair, (const float*)d_in[28],
                                               (const float*)d_in[29], pair_ln, PD);
    launch_gemm(stream, 2, 0, 1, 0, pair_ln, PD, wtq_i, PD, (const float*)d_in[31], nullptr, qb, 32, PR, 32, 32, PD);
    launch_gemm(stream, 2, 0, 1, 0, pair_ln, PD, wtk_i, PD, (const float*)d_in[33], nullptr, kb, 32, PR, 32, 32, PD);
    launch_gemm(stream, 2, 0, 2, 0, pair_ln, PD, wtv_i, PD, (const float*)d_in[35], nullptr, vb, 32, PR, 32, 32, PD);
    launch_gemm(stream, 2, 2, 0, 0, pair_ln, PD, wtg_i, PD, (const float*)d_in[39], gate_f, nullptr, 32, PR, 32, 32, PD);
    launch_gemm(stream, 1, 0, 0, 0, pair_ln, PD, wtb_i, PD, (const float*)d_in[37], bias_f, nullptr, 4, PR, 4, 16, PD);
    {
        dim3 g(L / 16, PH, L);
        attn_kernel<0><<<g, 32, 0, stream>>>(
            qb, kb, vb, bias_f, gate_f, attn_o, nullptr, L, PC, (int)PR,
            (long)L * 32, 32, 8, (long)L * 32, 32, 8,
            (long)L * PH, PH, 1, (long)L * 32, 32, 8, (long)L * 32, 32, 8, iscale, 0);
    }
    launch_gemm(stream, 4, 0, 0, 1, attn_o, 32, wto_i, 32, (const float*)d_in[41], outPair,
                nullptr, PD, PR, PD, PD, 32);

    // ================= Stage 5: pair attention (terminal / column-wise) ======
    ln_bf16_kernel<<<(int)PR, PD, 0, stream>>>(outPair, (const float*)d_in[42],
                                               (const float*)d_in[43], pair_ln, PD);
    launch_gemm(stream, 2, 0, 1, 0, pair_ln, PD, wtq_t, PD, (const float*)d_in[45], nullptr, qb, 32, PR, 32, 32, PD);
    launch_gemm(stream, 2, 0, 1, 0, pair_ln, PD, wtk_t, PD, (const float*)d_in[47], nullptr, kb, 32, PR, 32, 32, PD);
    launch_gemm(stream, 2, 0, 2, 0, pair_ln, PD, wtv_t, PD, (const float*)d_in[49], nullptr, vb, 32, PR, 32, 32, PD);
    launch_gemm(stream, 2, 2, 0, 0, pair_ln, PD, wtg_t, PD, (const float*)d_in[53], gate_f, nullptr, 32, PR, 32, 32, PD);
    launch_gemm(stream, 1, 0, 0, 0, pair_ln, PD, wtb_t, PD, (const float*)d_in[51], bias_f, nullptr, 4, PR, 4, 16, PD);
    for (int h = 0; h < PH; ++h) {
        dim3 g1(L / 16, 1, L);
        // logits[b,l,k] = q[b,l]·k[k,b]·s + bias[k,l];  K batch/key strides swapped,
        // bias query/key strides swapped.
        attn_kernel<1><<<g1, 32, 0, stream>>>(
            qb, kb, vb, bias_f, gate_f, nullptr, probs, L, PC, (int)PR,
            (long)L * 32, 32, 8,           // q: [b,l,h,c]
            32, (long)L * 32, 8,           // k: [k,b,h,c]
            PH, (long)L * PH, 1,           // bias[k,l,h]: per-query=PH, per-key=L*PH
            (long)L * 32, 32, 8, (long)L * 32, 32, 8, iscale, h);
        attn_apply_t_kernel<<<L, 256, 0, stream>>>(probs, vb, gate_f, attn_o, L, PH * PC, PC,
                                                   h, (int)PR);
    }
    launch_gemm(stream, 4, 0, 0, 1, attn_o, 32, wto_t, 32, (const float*)d_in[55], outPair,
                nullptr, PD, PR, PD, PD, 32);

    // ================= Stage 6: pair transition =================
    ln_bf16_kernel<<<(int)PR, PD, 0, stream>>>(outPair, (const float*)d_in[56],
                                               (const float*)d_in[57], pair_ln, PD);
    launch_gemm(stream, 4, 1, 1, 0, pair_ln, PD, wt1_p, PD, (const float*)d_in[59], nullptr, h1,
                EXP * PD, PR, EXP * PD, EXP * PD, PD);
    launch_gemm(stream, 4, 0, 0, 1, h1, EXP * PD, wt2_p, EXP * PD, (const float*)d_in[61],
                outPair, nullptr, PD, PR, PD, PD, EXP * PD);
}
